// WindowMHSA_46420006535712
// MI455X (gfx1250) — matrix operand order repaired and checked by reading, not compiled, not run on hardware
//
#include <hip/hip_runtime.h>

typedef _Float16 half8 __attribute__((ext_vector_type(8)));
typedef _Float16 v16h  __attribute__((ext_vector_type(16)));
typedef float    v8f   __attribute__((ext_vector_type(8)));

#define C_DIM   256
#define HEADS   8
#define SCALE_F 0.17677669529663687f   // 32^-0.5

// ---- LDS layout (units: f16 halves) ----
#define XSTR 264                        // xw[token][c], padded
#define QSTR 40                         // q/k [token][d], padded
#define VSTR 72                         // v  [d][token], padded
#define PSTR 72                         // P  [m][n], padded
#define OSTR 264                        // obuf[token][c], padded
#define XW_OFF   0
#define XW_SZ    (64 * XSTR)            // 16896
#define WAVE_OFF XW_SZ
#define PW_Q 0
#define PW_K (64 * QSTR)                // 2560
#define PW_V (2 * 64 * QSTR)            // 5120
#define PW_P (2 * 64 * QSTR + 32 * VSTR)// 7424
#define PW_SZ (PW_P + 64 * PSTR)        // 12032 halves / wave
#define OB_OFF (WAVE_OFF + HEADS * PW_SZ)
#define SMEM_HALVES (OB_OFF + 64 * OSTR)
#define SMEM_BYTES  (SMEM_HALVES * 2)   // 260096 B < 320KB/WGP

static __device__ __forceinline__ v16h mk16(const _Float16* p0, const _Float16* p1) {
  half8 lo = *(const half8*)p0;
  half8 hi = *(const half8*)p1;
  v16h r;
#pragma unroll
  for (int i = 0; i < 8; ++i) { r[i] = lo[i]; r[i + 8] = hi[i]; }
  return r;
}

#define WMMA(Afrag, Bfrag, Cacc) \
  __builtin_amdgcn_wmma_f32_16x16x32_f16(false, (Afrag), false, (Bfrag), (short)0, (Cacc), false, false)

// ---------------- weight fp32 -> fp16 convert (runs once per launch) ----------------
__global__ void wconvert_k(const float* __restrict__ qkvw,
                           const float* __restrict__ projw,
                           _Float16* __restrict__ dst) {
  int base = (blockIdx.x * blockDim.x + threadIdx.x) * 4;
#pragma unroll
  for (int j = 0; j < 4; ++j) {
    int i = base + j;                       // 0 .. 262143
    if (i < 768 * 256) dst[i] = (_Float16)qkvw[i];
    else               dst[i] = (_Float16)projw[i - 768 * 256];
  }
}

// ---------------- fused per-window MHSA ----------------
__global__ __launch_bounds__(256, 1)
void win_mhsa_k(const float* __restrict__ x,
                const float* __restrict__ qkv_b,
                const float* __restrict__ proj_b,
                const _Float16* __restrict__ wh,   // f16 weights in workspace
                float* __restrict__ out) {
  extern __shared__ _Float16 smem[];
  const _Float16* qkvw_h = wh;
  const _Float16* projw_h = wh + 768 * 256;

  const int tid  = threadIdx.x;
  const int lane = tid & 31;
  const int wv   = tid >> 5;        // wave id == head id
  const int l16  = lane & 15;
  const int g    = lane >> 4;       // half-wave group

  const int w  = blockIdx.x;        // window id, 2048 total
  const int b  = w >> 6;
  const int wy = (w >> 3) & 7;
  const int wx = w & 7;

  // ---------------- Stage A: window load, f32->f16, LDS xw[token][c] ----------------
  {
    const float* xg = x + (size_t)b * (256 * 64 * 64);
    for (int i = tid; i < 2048; i += 256) {   // (c, py) pairs, exact 8 iters, no divergence
      int c = i >> 3, py = i & 7;
      const float4* s4 = (const float4*)(xg + ((size_t)c * 64 + (wy * 8 + py)) * 64 + wx * 8);
      float4 u = s4[0], v2 = s4[1];
      _Float16* col = smem + XW_OFF + c;
      int pb0 = py * 8;
      col[(pb0 + 0) * XSTR] = (_Float16)u.x;  col[(pb0 + 1) * XSTR] = (_Float16)u.y;
      col[(pb0 + 2) * XSTR] = (_Float16)u.z;  col[(pb0 + 3) * XSTR] = (_Float16)u.w;
      col[(pb0 + 4) * XSTR] = (_Float16)v2.x; col[(pb0 + 5) * XSTR] = (_Float16)v2.y;
      col[(pb0 + 6) * XSTR] = (_Float16)v2.z; col[(pb0 + 7) * XSTR] = (_Float16)v2.w;
    }
  }
  __syncthreads();

  _Float16* qb = smem + WAVE_OFF + wv * PW_SZ + PW_Q;  // [token][d]  (holds SCALE*q)
  _Float16* kb = smem + WAVE_OFF + wv * PW_SZ + PW_K;  // [token][d]
  _Float16* vb = smem + WAVE_OFF + wv * PW_SZ + PW_V;  // [d][token]
  _Float16* pbuf = smem + WAVE_OFF + wv * PW_SZ + PW_P;// [m][n]
  _Float16* ob = smem + OB_OFF;                        // [token][c], shared

  // ---------------- Stage B: QKV GEMM, register-blocked (B frags loaded once per k) ----
  {
    // A row pointers for the 6 output tile-rows (q0,q1,k0,k1,v0,v1)
    const _Float16* arow[6];
#pragma unroll
    for (int tr = 0; tr < 6; ++tr) {
      const int rowbase = (tr >> 1) * 256 + wv * 32 + (tr & 1) * 16;
      arow[tr] = qkvw_h + (size_t)(rowbase + l16) * 256;
    }

    v8f acc[6][4] = {};
    for (int k = 0; k < 256; k += 32) {
      v16h bf[4];
#pragma unroll
      for (int tc = 0; tc < 4; ++tc) {
        const _Float16* br = smem + XW_OFF + (size_t)(tc * 16 + l16) * XSTR + k + g * 8;
        bf[tc] = mk16(br, br + 16);
      }
#pragma unroll
      for (int tr = 0; tr < 6; ++tr) {
        v16h a = mk16(arow[tr] + k + g * 8, arow[tr] + k + 16 + g * 8);
#pragma unroll
        for (int tc = 0; tc < 4; ++tc)
          acc[tr][tc] = WMMA(a, bf[tc], acc[tr][tc]);
      }
    }

#pragma unroll
    for (int tr = 0; tr < 6; ++tr) {
      const int sect = tr >> 1;                               // 0=q 1=k 2=v
      const int rowbase = sect * 256 + wv * 32 + (tr & 1) * 16;
#pragma unroll
      for (int tc = 0; tc < 4; ++tc) {
#pragma unroll
        for (int r = 0; r < 8; ++r) {
          int m = r + 8 * g;                       // row within 16-tile
          float f = acc[tr][tc][r] + qkv_b[rowbase + m];
          int t = tc * 16 + l16;                   // token
          int s = (tr & 1) * 16 + m;               // d within head
          if (sect == 0)      qb[t * QSTR + s] = (_Float16)(f * SCALE_F); // fold softmax scale
          else if (sect == 1) kb[t * QSTR + s] = (_Float16)f;
          else                vb[s * VSTR + t] = (_Float16)f;
        }
      }
    }
  }

  // ---------------- Stage C: S = (SCALE*Q) K^T (d=32 is exactly one WMMA K-step) -------
  v8f S[4][4] = {};
#pragma unroll
  for (int ti = 0; ti < 4; ++ti) {
    const _Float16* qr = qb + (ti * 16 + l16) * QSTR;
    v16h qa = mk16(qr + g * 8, qr + 16 + g * 8);
#pragma unroll
    for (int tj = 0; tj < 4; ++tj) {
      const _Float16* kr = kb + (tj * 16 + l16) * QSTR + g * 8;
      v16h kf = mk16(kr, kr + 16);
      S[ti][tj] = WMMA(qa, kf, S[ti][tj]);
    }
  }

  // ---------------- softmax in registers (row = r + 8g; 16 lanes hold 16 cols) ---------
#pragma unroll
  for (int ti = 0; ti < 4; ++ti) {
#pragma unroll
    for (int r = 0; r < 8; ++r) {
      float v0 = S[ti][0][r], v1 = S[ti][1][r];
      float v2 = S[ti][2][r], v3 = S[ti][3][r];
      float mx = fmaxf(fmaxf(v0, v1), fmaxf(v2, v3));
      mx = fmaxf(mx, __shfl_xor(mx, 1, 32));
      mx = fmaxf(mx, __shfl_xor(mx, 2, 32));
      mx = fmaxf(mx, __shfl_xor(mx, 4, 32));
      mx = fmaxf(mx, __shfl_xor(mx, 8, 32));
      float e0 = __expf(v0 - mx), e1 = __expf(v1 - mx);
      float e2 = __expf(v2 - mx), e3 = __expf(v3 - mx);
      float s = e0 + e1 + e2 + e3;
      s += __shfl_xor(s, 1, 32); s += __shfl_xor(s, 2, 32);
      s += __shfl_xor(s, 4, 32); s += __shfl_xor(s, 8, 32);
      float inv = 1.0f / s;
      int mrow = ti * 16 + r + 8 * g;
      _Float16* pr = pbuf + mrow * PSTR + l16;
      pr[0]  = (_Float16)(e0 * inv);
      pr[16] = (_Float16)(e1 * inv);
      pr[32] = (_Float16)(e2 * inv);
      pr[48] = (_Float16)(e3 * inv);
    }
  }

  // ---------------- Stage D: O = P V ----------------
  v8f O[4][2] = {};
#pragma unroll
  for (int tk = 0; tk < 2; ++tk) {
#pragma unroll
    for (int dj = 0; dj < 2; ++dj) {
      const _Float16* vr = vb + (dj * 16 + l16) * VSTR + tk * 32 + g * 8;
      v16h vf = mk16(vr, vr + 16);
#pragma unroll
      for (int ti = 0; ti < 4; ++ti) {
        const _Float16* prr = pbuf + (ti * 16 + l16) * PSTR + tk * 32;
        v16h pa = mk16(prr + g * 8, prr + 16 + g * 8);
        O[ti][dj] = WMMA(pa, vf, O[ti][dj]);
      }
    }
  }
#pragma unroll
  for (int ti = 0; ti < 4; ++ti)
#pragma unroll
    for (int dj = 0; dj < 2; ++dj)
#pragma unroll
      for (int r = 0; r < 8; ++r) {
        int t = ti * 16 + r + 8 * g;
        int c = wv * 32 + dj * 16 + l16;
        ob[t * OSTR + c] = (_Float16)O[ti][dj][r];
      }

  __syncthreads();

  // ---------------- Stage E: proj GEMM, register-blocked, bias, fp32 store -------------
  {
    const int o0 = wv * 32;
    const _Float16* arow0 = projw_h + (size_t)(o0 + l16) * 256;
    const _Float16* arow1 = projw_h + (size_t)(o0 + 16 + l16) * 256;

    v8f acc[2][4] = {};
    for (int k = 0; k < 256; k += 32) {
      v16h bf[4];
#pragma unroll
      for (int tc = 0; tc < 4; ++tc) {
        const _Float16* br = ob + (tc * 16 + l16) * OSTR + k + g * 8;
        bf[tc] = mk16(br, br + 16);
      }
      v16h a0 = mk16(arow0 + k + g * 8, arow0 + k + 16 + g * 8);
      v16h a1 = mk16(arow1 + k + g * 8, arow1 + k + 16 + g * 8);
#pragma unroll
      for (int tc = 0; tc < 4; ++tc) {
        acc[0][tc] = WMMA(a0, bf[tc], acc[0][tc]);
        acc[1][tc] = WMMA(a1, bf[tc], acc[1][tc]);
      }
    }

#pragma unroll
    for (int tr = 0; tr < 2; ++tr) {
#pragma unroll
      for (int tc = 0; tc < 4; ++tc) {
#pragma unroll
        for (int r = 0; r < 8; ++r) {
          int o = o0 + tr * 16 + r + 8 * g;
          float f = acc[tr][tc][r] + proj_b[o];
          int p = tc * 16 + l16, py = p >> 3, px = p & 7;
          out[(((size_t)b * 256 + o) * 64 + wy * 8 + py) * 64 + wx * 8 + px] = f;
        }
      }
    }
  }
}

extern "C" void kernel_launch(void* const* d_in, const int* in_sizes, int n_in,
                              void* d_out, int out_size, void* d_ws, size_t ws_size,
                              hipStream_t stream) {
  const float* x     = (const float*)d_in[0];
  const float* qkvw  = (const float*)d_in[1];
  const float* qkvb  = (const float*)d_in[2];
  const float* projw = (const float*)d_in[3];
  const float* projb = (const float*)d_in[4];
  _Float16* wh = (_Float16*)d_ws;   // 768*256 + 256*256 halves ~= 524KB in d_ws

  (void)in_sizes; (void)n_in; (void)out_size; (void)ws_size;

  hipFuncSetAttribute(reinterpret_cast<const void*>(win_mhsa_k),
                      hipFuncAttributeMaxDynamicSharedMemorySize, SMEM_BYTES);

  wconvert_k<<<256, 256, 0, stream>>>(qkvw, projw, wh);
  win_mhsa_k<<<2048, 256, SMEM_BYTES, stream>>>(x, qkvb, projb, wh, (float*)d_out);
}
